// RelativeMultiHeadAttention_18219251269899
// MI455X (gfx1250) — matrix operand order repaired and checked by reading, not compiled, not run on hardware
//
#include <hip/hip_runtime.h>
#include <hip/hip_bf16.h>

// ---------------------------------------------------------------------------
// Relative Multi-Head Attention for MI455X (gfx1250, wave32, WMMA + TDM).
// B=4, S=1024, D=1024, H=16, hd=64, M=128 (rel table 257 rows, padded to 288).
//
// Pipeline:
//   1) cvt:     f32 -> bf16 for x, Wq, Wk, Wv, Wo
//   2) relprep: rel_k -> bf16 [288][64] (zero-padded), rel_v -> bf16^T [64][288]
//   3) gemm_bias<bf16 out> x3:   q/k/v = x @ W^T + b      (WMMA, TDM-staged)
//   4) attn:    fused relative attention per (b,h,qblock) (WMMA, TDM-staged K)
//   5) gemm_bias<f32 out>:       out = attn @ Wo^T + bo   (WMMA, TDM-staged)
// ---------------------------------------------------------------------------

typedef __bf16 bf16;
typedef bf16  v16bf __attribute__((ext_vector_type(16)));
typedef float v8f   __attribute__((ext_vector_type(8)));

#ifndef __has_builtin
#define __has_builtin(x) 0
#endif
#if __has_builtin(__builtin_amdgcn_tensor_load_to_lds) && \
    __has_builtin(__builtin_amdgcn_s_wait_tensorcnt)
#define HAS_TDM 1
#else
#define HAS_TDM 0
#endif

__device__ __forceinline__ bf16 f2bf(float f) {
  union { float f; unsigned u; } a; a.f = f;
  unsigned r = a.u + 0x7FFFu + ((a.u >> 16) & 1u);   // round-to-nearest-even
  unsigned short h = (unsigned short)(r >> 16);
  return __builtin_bit_cast(bf16, h);
}

__device__ __forceinline__ v8f wmma_bf16(v16bf a, v16bf b, v8f c) {
  // D = A(16x32 bf16) x B(32x16 bf16) + C(16x16 f32)
  return __builtin_amdgcn_wmma_f32_16x16x32_bf16(
      /*neg_a=*/false, a, /*neg_b=*/false, b,
      /*c_mod=*/(short)0, c, /*reuse_a=*/false, /*reuse_b=*/false);
}

#if HAS_TDM
typedef unsigned int u32x4 __attribute__((ext_vector_type(4)));
typedef int          i32x8 __attribute__((ext_vector_type(8)));
typedef int          i32x4 __attribute__((ext_vector_type(4)));

// Issue one TDM 2-D tile load (bf16 elements) global -> LDS, per D# spec
// (CDNA5 ISA). Tile is stored contiguous row-major at
// lds_dst; global rows are row_stride elements apart. Tracked by TENSORcnt.
__device__ __forceinline__ void tdm_load_tile_2d(void* lds_dst, const void* gsrc,
                                                 unsigned tile_d0, unsigned tile_d1,
                                                 unsigned long long row_stride) {
  unsigned lds_off = (unsigned)(unsigned long long)lds_dst;   // addr[31:0] = LDS offset
  unsigned long long ga = (unsigned long long)gsrc;
  u32x4 g0;
  g0[0] = 1u;                                          // count=1, user descriptor
  g0[1] = lds_off;                                     // lds_addr
  g0[2] = (unsigned)(ga & 0xFFFFFFFFu);                // global_addr[31:0]
  g0[3] = (unsigned)((ga >> 32) & 0x01FFFFFFu)         // global_addr[56:32]
        | (2u << 30);                                  // type = 2 ("image")
  i32x8 g1;
  g1[0] = (int)(1u << 16);                             // data_size=1 (2 bytes)
  g1[1] = (int)((tile_d0 & 0xFFFFu) << 16);            // tensor_dim0[15:0] @63:48
  g1[2] = (int)(((tile_d0 >> 16) & 0xFFFFu)            // tensor_dim0[31:16]
        | ((tile_d1 & 0xFFFFu) << 16));                // tensor_dim1[15:0]
  g1[3] = (int)(((tile_d1 >> 16) & 0xFFFFu)            // tensor_dim1[31:16]
        | ((tile_d0 & 0xFFFFu) << 16));                // tile_dim0 @127:112
  g1[4] = (int)(tile_d1 & 0xFFFFu);                    // tile_dim1 (tile_dim2=0)
  g1[5] = (int)(unsigned)(row_stride & 0xFFFFFFFFull); // tensor_dim0_stride[31:0]
  g1[6] = (int)(unsigned)((row_stride >> 32) & 0xFFFFull); // stride[47:32]
  g1[7] = 0;                                           // tensor_dim1_stride = 0
  i32x4 z4 = {0, 0, 0, 0};
#if __clang_major__ >= 23
  i32x8 z8 = {0, 0, 0, 0, 0, 0, 0, 0};
  __builtin_amdgcn_tensor_load_to_lds(g0, g1, z4, z4, z8, 0);
#else
  __builtin_amdgcn_tensor_load_to_lds(g0, g1, z4, z4, 0);
#endif
}
#endif // HAS_TDM

// A fragment, 16x32 bf16, row-major source with leading dim `ld` (elements).
__device__ __forceinline__ v16bf load_frag_a(const bf16* base, int ld) {
  int lane = threadIdx.x & 31;
  int half = lane >> 4;
  int r    = lane & 15;
  const bf16* p = base + (size_t)r * ld + half * 8;
  union { v16bf v; uint4 q[2]; } u;
  u.q[0] = *(const uint4*)(p);        // K = kbase .. kbase+7
  u.q[1] = *(const uint4*)(p + 16);   // K = kbase+16 .. kbase+23
  return u.v;
}

// B fragment, 32x16 bf16 (K x N). Source stored N-major (row n = column n of B).
__device__ __forceinline__ v16bf load_frag_b(const bf16* base, int ld) {
  int lane = threadIdx.x & 31;
  int half = lane >> 4;
  int n    = lane & 15;
  const bf16* p = base + (size_t)n * ld + half * 8;
  union { v16bf v; uint4 q[2]; } u;
  u.q[0] = *(const uint4*)(p);
  u.q[1] = *(const uint4*)(p + 16);
  return u.v;
}

// A fragment built from an f32 source (converted on the fly).
__device__ __forceinline__ v16bf load_frag_a_f32(const float* base, int ld) {
  int lane = threadIdx.x & 31;
  int half = lane >> 4;
  int r    = lane & 15;
  const float* p = base + (size_t)r * ld + half * 8;
  v16bf v;
#pragma unroll
  for (int i = 0; i < 8; ++i) {
    v[i]     = f2bf(p[i]);
    v[i + 8] = f2bf(p[i + 16]);
  }
  return v;
}

// ---------------------------------------------------------------------------
// Kernel 1: elementwise f32 -> bf16
// ---------------------------------------------------------------------------
__global__ void cvt_kernel(const float* __restrict__ s, bf16* __restrict__ d, int n) {
  int i = blockIdx.x * blockDim.x + threadIdx.x;
  if (i < n) d[i] = f2bf(s[i]);
}

// ---------------------------------------------------------------------------
// Kernel 2: rel tables. relkb: [288][64] bf16, relvT: [64][288] bf16.
// ---------------------------------------------------------------------------
__global__ void relprep_kernel(const float* __restrict__ rk, const float* __restrict__ rv,
                               bf16* __restrict__ relkb, bf16* __restrict__ relvT) {
  int i = blockIdx.x * blockDim.x + threadIdx.x;
  if (i >= 288 * 64) return;
  int j = i >> 6, d = i & 63;
  float vk = (j < 257) ? rk[j * 64 + d] : 0.f;
  float vv = (j < 257) ? rv[j * 64 + d] : 0.f;
  relkb[j * 64 + d]  = f2bf(vk);
  relvT[d * 288 + j] = f2bf(vv);
}

// ---------------------------------------------------------------------------
// Kernel 3/5: C[M][N] = A[M][K] @ B[N][K]^T + bias.  64x64 tile, 128 threads
// (4 waves, 16-row strip each), K stepped by 32, double-buffered TDM staging.
// ---------------------------------------------------------------------------
template <int OUT_BF16>
__global__ __launch_bounds__(128)
void gemm_bias_kernel(const bf16* __restrict__ A, const bf16* __restrict__ Bw,
                      const float* __restrict__ bias, void* __restrict__ Out,
                      int Mdim, int Ndim, int Kdim) {
  __shared__ bf16 As[2][64 * 32];
  __shared__ bf16 Bs[2][64 * 32];
  int tid  = threadIdx.x;
  int wave = tid >> 5, lane = tid & 31;
  int half = lane >> 4, cn = lane & 15;
  int m0 = blockIdx.x * 64, n0 = blockIdx.y * 64;

  v8f acc[4] = {v8f{}, v8f{}, v8f{}, v8f{}};

#if HAS_TDM
  // prologue: DMA first k-step tiles, wave 0 drives the Tensor Data Mover
  if (wave == 0) {
    tdm_load_tile_2d(As[0], A  + (size_t)m0 * Kdim, 32, 64, (unsigned long long)Kdim);
    tdm_load_tile_2d(Bs[0], Bw + (size_t)n0 * Kdim, 32, 64, (unsigned long long)Kdim);
    __builtin_amdgcn_s_wait_tensorcnt(0);
  }
  __syncthreads();
  for (int k0 = 0; k0 < Kdim; k0 += 32) {
    int cur = (k0 >> 5) & 1;
    if (wave == 0 && k0 + 32 < Kdim) {   // overlap next DMA with current WMMAs
      tdm_load_tile_2d(As[cur ^ 1], A  + (size_t)m0 * Kdim + k0 + 32, 32, 64,
                       (unsigned long long)Kdim);
      tdm_load_tile_2d(Bs[cur ^ 1], Bw + (size_t)n0 * Kdim + k0 + 32, 32, 64,
                       (unsigned long long)Kdim);
    }
    __builtin_prefetch(A + (size_t)(m0 + (tid & 63)) * Kdim + k0 + 64, 0, 1);
    v16bf a = load_frag_a(As[cur] + wave * 16 * 32, 32);
#pragma unroll
    for (int t = 0; t < 4; ++t) {
      v16bf b = load_frag_b(Bs[cur] + t * 16 * 32, 32);
      acc[t] = wmma_bf16(a, b, acc[t]);
    }
    if (wave == 0) __builtin_amdgcn_s_wait_tensorcnt(0);
    __syncthreads();
  }
#else
  for (int k0 = 0; k0 < Kdim; k0 += 32) {
    int idx = tid * 16;
    int r = idx >> 5, c = idx & 31;
    const bf16* ga = A  + (size_t)(m0 + r) * Kdim + k0 + c;
    const bf16* gb = Bw + (size_t)(n0 + r) * Kdim + k0 + c;
    *(uint4*)(As[0] + idx)     = *(const uint4*)(ga);
    *(uint4*)(As[0] + idx + 8) = *(const uint4*)(ga + 8);
    *(uint4*)(Bs[0] + idx)     = *(const uint4*)(gb);
    *(uint4*)(Bs[0] + idx + 8) = *(const uint4*)(gb + 8);
    __syncthreads();
    v16bf a = load_frag_a(As[0] + wave * 16 * 32, 32);
#pragma unroll
    for (int t = 0; t < 4; ++t) {
      v16bf b = load_frag_b(Bs[0] + t * 16 * 32, 32);
      acc[t] = wmma_bf16(a, b, acc[t]);
    }
    __syncthreads();
  }
#endif

#pragma unroll
  for (int t = 0; t < 4; ++t) {
    int col = n0 + t * 16 + cn;
    float bvv = bias ? bias[col] : 0.f;
#pragma unroll
    for (int j = 0; j < 8; ++j) {
      int row = m0 + wave * 16 + half * 8 + j;
      float val = acc[t][j] + bvv;
      if (OUT_BF16) ((bf16*)Out)[(size_t)row * Ndim + col] = f2bf(val);
      else          ((float*)Out)[(size_t)row * Ndim + col] = val;
    }
  }
  (void)Mdim;
}

// ---------------------------------------------------------------------------
// Kernel 4: fused relative attention.
// grid = (S/32, B*H); block = 64 threads (2 waves, 16 query rows each).
//   qrel[32][288] = (Q @ rel_k^T) * scale        (WMMA)
//   pass1: row-max over all S (scores = QK^T*scale + qrel gather), TDM K tiles
//   pass2: p=exp(s-m); l+=p; wbar[q][idx]+=p (ds_add); O += P@V (WMMA)
//   O += wbar @ rel_v^T (WMMA, K=288);  out = O / l
// ---------------------------------------------------------------------------
__global__ __launch_bounds__(64)
void attn_kernel(const bf16* __restrict__ qb, const bf16* __restrict__ kb,
                 const bf16* __restrict__ vb, const bf16* __restrict__ relkb,
                 const bf16* __restrict__ relvT, bf16* __restrict__ attnb,
                 int Bn, int S, int H, int hd) {
  const int QB = 32, KB = 64, RJ = 288;
  extern __shared__ char smem[];
  bf16*  Qs   = (bf16*)smem;                 // [32][64]
  bf16*  Ks   = Qs + QB * 64;                // [2][64][64] double-buffered
  bf16*  Vs   = Ks + 2 * KB * 64;            // [64][64]  transposed: [d][k]
  bf16*  Ps   = Vs + 64 * 64;                // [32][64]
  float* qrel = (float*)(Ps + QB * 64);      // [32][288]
  float* wbar = qrel + QB * RJ;              // [32][288]

  int tid  = threadIdx.x;
  int wave = tid >> 5, lane = tid & 31;
  int half = lane >> 4, ln = lane & 15;
  int q0 = blockIdx.x * QB;
  int bh = blockIdx.y;
  int b = bh / H, h = bh % H;
  size_t D = (size_t)H * hd;

  const bf16* qhead = qb + (size_t)b * S * D + (size_t)h * hd;
  const bf16* khead = kb + (size_t)b * S * D + (size_t)h * hd;
  const bf16* vhead = vb + (size_t)b * S * D + (size_t)h * hd;

  // stage Q strip (TDM when available), zero wbar
#if HAS_TDM
  if (wave == 0)
    tdm_load_tile_2d(Qs, qhead + (size_t)q0 * D, 64, QB, (unsigned long long)D);
#else
  for (int i = tid; i < QB * 64 / 8; i += 64) {
    int e = i * 8; int r = e >> 6, c = e & 63;
    *(uint4*)(Qs + e) = *(const uint4*)(qhead + (size_t)(q0 + r) * D + c);
  }
#endif
  for (int i = tid; i < QB * RJ; i += 64) wbar[i] = 0.f;
#if HAS_TDM
  if (wave == 0) __builtin_amdgcn_s_wait_tensorcnt(0);
#endif
  __syncthreads();

  // Q fragments for this wave's 16 rows (K = hd = 64 -> two chunks)
  const bf16* Qw = Qs + wave * 16 * 64;
  v16bf qa0 = load_frag_a(Qw, 64);
  v16bf qa1 = load_frag_a(Qw + 32, 64);

  // qrel = (Q @ rel_k^T) * scale, 18 column tiles of 16
  const float scale = 0.125f; // 1/sqrt(64)
  for (int t = 0; t < 18; ++t) {
    v16bf b0 = load_frag_b(relkb + (size_t)(t * 16) * 64, 64);
    v16bf b1 = load_frag_b(relkb + (size_t)(t * 16) * 64 + 32, 64);
    v8f c = {};
    c = wmma_bf16(qa0, b0, c);
    c = wmma_bf16(qa1, b1, c);
#pragma unroll
    for (int j = 0; j < 8; ++j) {
      int rl = wave * 16 + half * 8 + j;
      qrel[rl * RJ + t * 16 + ln] = c[j] * scale;
    }
  }
  __syncthreads();

  // ---------------- pass 1: row max ----------------
  float m[8];
#pragma unroll
  for (int j = 0; j < 8; ++j) m[j] = -1e30f;

#if HAS_TDM
  if (wave == 0) {
    tdm_load_tile_2d(Ks, khead, 64, KB, (unsigned long long)D);
    __builtin_amdgcn_s_wait_tensorcnt(0);
  }
  __syncthreads();
#endif
  for (int kb0 = 0; kb0 < S; kb0 += KB) {
#if HAS_TDM
    int cur = (kb0 >> 6) & 1;
    if (wave == 0 && kb0 + KB < S)   // prefetch next K tile via TDM
      tdm_load_tile_2d(Ks + (cur ^ 1) * KB * 64, khead + (size_t)(kb0 + KB) * D,
                       64, KB, (unsigned long long)D);
    const bf16* Kt = Ks + cur * KB * 64;
#else
    const bf16* Kt = Ks;
    for (int i = tid; i < KB * 64 / 8; i += 64) {
      int e = i * 8; int r = e >> 6, c = e & 63;
      *(uint4*)(Ks + e) = *(const uint4*)(khead + (size_t)(kb0 + r) * D + c);
    }
    __syncthreads();
#endif
#pragma unroll
    for (int t = 0; t < 4; ++t) {
      v16bf b0 = load_frag_b(Kt + (size_t)(t * 16) * 64, 64);
      v16bf b1 = load_frag_b(Kt + (size_t)(t * 16) * 64 + 32, 64);
      v8f c = {};
      c = wmma_bf16(qa0, b0, c);
      c = wmma_bf16(qa1, b1, c);
#pragma unroll
      for (int j = 0; j < 8; ++j) {
        int rl = wave * 16 + half * 8 + j;
        int qrow = q0 + rl;
        int kcol = kb0 + t * 16 + ln;
        int rel = kcol - qrow;
        rel = rel < -128 ? -128 : (rel > 128 ? 128 : rel);
        float s = c[j] * scale + qrel[rl * RJ + rel + 128];
        m[j] = fmaxf(m[j], s);
      }
    }
#if HAS_TDM
    if (wave == 0) __builtin_amdgcn_s_wait_tensorcnt(0);
#endif
    __syncthreads();
  }
#pragma unroll
  for (int j = 0; j < 8; ++j)
    for (int mk = 8; mk >= 1; mk >>= 1)
      m[j] = fmaxf(m[j], __shfl_xor(m[j], mk, 32));

  // ---------------- pass 2: softmax + P@V + wbar ----------------
  float l[8];
#pragma unroll
  for (int j = 0; j < 8; ++j) l[j] = 0.f;
  v8f acc[4] = {v8f{}, v8f{}, v8f{}, v8f{}};

#if HAS_TDM
  if (wave == 0) {
    tdm_load_tile_2d(Ks, khead, 64, KB, (unsigned long long)D);
    __builtin_amdgcn_s_wait_tensorcnt(0);
  }
  __syncthreads();
#endif
  for (int kb0 = 0; kb0 < S; kb0 += KB) {
#if HAS_TDM
    int cur = (kb0 >> 6) & 1;
    if (wave == 0 && kb0 + KB < S)
      tdm_load_tile_2d(Ks + (cur ^ 1) * KB * 64, khead + (size_t)(kb0 + KB) * D,
                       64, KB, (unsigned long long)D);
    const bf16* Kt = Ks + cur * KB * 64;
    // V: load 8 contiguous d, scatter-transpose into Vs[d][k]
    for (int i = tid; i < KB * 64 / 8; i += 64) {
      int e = i * 8; int r = e >> 6, c = e & 63;
      union { uint4 u; bf16 x[8]; } tv;
      tv.u = *(const uint4*)(vhead + (size_t)(kb0 + r) * D + c);
#pragma unroll
      for (int z = 0; z < 8; ++z) Vs[(c + z) * 64 + r] = tv.x[z];
    }
    __syncthreads();
#else
    const bf16* Kt = Ks;
    for (int i = tid; i < KB * 64 / 8; i += 64) {
      int e = i * 8; int r = e >> 6, c = e & 63;
      *(uint4*)(Ks + e) = *(const uint4*)(khead + (size_t)(kb0 + r) * D + c);
      union { uint4 u; bf16 x[8]; } tv;
      tv.u = *(const uint4*)(vhead + (size_t)(kb0 + r) * D + c);
#pragma unroll
      for (int z = 0; z < 8; ++z) Vs[(c + z) * 64 + r] = tv.x[z];
    }
    __syncthreads();
#endif

#pragma unroll
    for (int t = 0; t < 4; ++t) {
      v16bf b0 = load_frag_b(Kt + (size_t)(t * 16) * 64, 64);
      v16bf b1 = load_frag_b(Kt + (size_t)(t * 16) * 64 + 32, 64);
      v8f c = {};
      c = wmma_bf16(qa0, b0, c);
      c = wmma_bf16(qa1, b1, c);
#pragma unroll
      for (int j = 0; j < 8; ++j) {
        int rl = wave * 16 + half * 8 + j;
        int qrow = q0 + rl;
        int kcol = kb0 + t * 16 + ln;
        int rel = kcol - qrow;
        rel = rel < -128 ? -128 : (rel > 128 ? 128 : rel);
        float s = c[j] * scale + qrel[rl * RJ + rel + 128];
        float p = __expf(s - m[j]);
        l[j] += p;
        atomicAdd(&wbar[rl * RJ + rel + 128], p);   // ds_add_f32
        Ps[rl * 64 + t * 16 + ln] = f2bf(p);
      }
    }
    __syncthreads();

    // O += P @ V
    const bf16* Pw = Ps + wave * 16 * 64;
    v16bf pa0 = load_frag_a(Pw, 64);
    v16bf pa1 = load_frag_a(Pw + 32, 64);
#pragma unroll
    for (int t = 0; t < 4; ++t) {
      v16bf b0 = load_frag_b(Vs + (size_t)(t * 16) * 64, 64);
      v16bf b1 = load_frag_b(Vs + (size_t)(t * 16) * 64 + 32, 64);
      acc[t] = wmma_bf16(pa0, b0, acc[t]);
      acc[t] = wmma_bf16(pa1, b1, acc[t]);
    }
#if HAS_TDM
    if (wave == 0) __builtin_amdgcn_s_wait_tensorcnt(0);
#endif
    __syncthreads();
  }

#pragma unroll
  for (int j = 0; j < 8; ++j)
    for (int mk = 8; mk >= 1; mk >>= 1)
      l[j] += __shfl_xor(l[j], mk, 32);
  __syncthreads();

  // O += wbar @ rel_v^T  (K = 288)
  const float* Ww = wbar + wave * 16 * RJ;
  for (int k0 = 0; k0 < RJ; k0 += 32) {
    v16bf a = load_frag_a_f32(Ww + k0, RJ);
#pragma unroll
    for (int t = 0; t < 4; ++t) {
      v16bf bfr = load_frag_b(relvT + (size_t)(t * 16) * RJ + k0, RJ);
      acc[t] = wmma_bf16(a, bfr, acc[t]);
    }
  }

  // write normalized output
  bf16* ohead = attnb + (size_t)b * S * D + (size_t)h * hd;
#pragma unroll
  for (int t = 0; t < 4; ++t) {
#pragma unroll
    for (int j = 0; j < 8; ++j) {
      int rl = wave * 16 + half * 8 + j;
      float inv = 1.f / l[j];
      ohead[(size_t)(q0 + rl) * D + t * 16 + ln] = f2bf(acc[t][j] * inv);
    }
  }
  (void)Bn;
}

// ---------------------------------------------------------------------------
extern "C" void kernel_launch(void* const* d_in, const int* in_sizes, int n_in,
                              void* d_out, int out_size, void* d_ws, size_t ws_size,
                              hipStream_t stream) {
  (void)in_sizes; (void)n_in; (void)out_size; (void)ws_size;
  const float* x     = (const float*)d_in[0];
  const float* Wq    = (const float*)d_in[1];
  const float* bq    = (const float*)d_in[2];
  const float* Wk    = (const float*)d_in[3];
  const float* bk    = (const float*)d_in[4];
  const float* Wv    = (const float*)d_in[5];
  const float* bv    = (const float*)d_in[6];
  const float* Wo    = (const float*)d_in[7];
  const float* bo    = (const float*)d_in[8];
  const float* rel_k = (const float*)d_in[9];
  const float* rel_v = (const float*)d_in[10];

  const int Bn = 4, S = 1024, D = 1024, H = 16;
  const int Mrows = Bn * S;  // 4096

  char* ws = (char*)d_ws;
  size_t off = 0;
  auto carve = [&](size_t bytes) -> char* {
    off = (off + 255) & ~(size_t)255;
    char* p = ws + off;
    off += bytes;
    return p;
  };
  bf16* xb    = (bf16*)carve((size_t)Mrows * D * 2);
  bf16* Wqb   = (bf16*)carve((size_t)D * D * 2);
  bf16* Wkb   = (bf16*)carve((size_t)D * D * 2);
  bf16* Wvb   = (bf16*)carve((size_t)D * D * 2);
  bf16* Wob   = (bf16*)carve((size_t)D * D * 2);
  bf16* qbuf  = (bf16*)carve((size_t)Mrows * D * 2);
  bf16* kbuf  = (bf16*)carve((size_t)Mrows * D * 2);
  bf16* vbuf  = (bf16*)carve((size_t)Mrows * D * 2);
  bf16* attnb = (bf16*)carve((size_t)Mrows * D * 2);
  bf16* relkb = (bf16*)carve((size_t)288 * 64 * 2);
  bf16* relvT = (bf16*)carve((size_t)64 * 288 * 2);

  // 1) convert inputs to bf16
  {
    int n = Mrows * D;
    cvt_kernel<<<(n + 255) / 256, 256, 0, stream>>>(x, xb, n);
    n = D * D;
    cvt_kernel<<<(n + 255) / 256, 256, 0, stream>>>(Wq, Wqb, n);
    cvt_kernel<<<(n + 255) / 256, 256, 0, stream>>>(Wk, Wkb, n);
    cvt_kernel<<<(n + 255) / 256, 256, 0, stream>>>(Wv, Wvb, n);
    cvt_kernel<<<(n + 255) / 256, 256, 0, stream>>>(Wo, Wob, n);
    int nr = 288 * 64;
    relprep_kernel<<<(nr + 255) / 256, 256, 0, stream>>>(rel_k, rel_v, relkb, relvT);
  }

  // 2) q/k/v projections (bf16 out)
  {
    dim3 grid(Mrows / 64, D / 64);
    gemm_bias_kernel<1><<<grid, 128, 0, stream>>>(xb, Wqb, bq, qbuf, Mrows, D, D);
    gemm_bias_kernel<1><<<grid, 128, 0, stream>>>(xb, Wkb, bk, kbuf, Mrows, D, D);
    gemm_bias_kernel<1><<<grid, 128, 0, stream>>>(xb, Wvb, bv, vbuf, Mrows, D, D);
  }

  // 3) fused relative attention
  {
    dim3 grid(S / 32, Bn * H);
    size_t sm = (size_t)(32 * 64 + 2 * 64 * 64 + 64 * 64 + 32 * 64) * 2  // Qs,Ks x2,Vs,Ps
              + (size_t)(32 * 288) * 4 * 2;                              // qrel + wbar
    attn_kernel<<<grid, 64, sm, stream>>>(qbuf, kbuf, vbuf, relkb, relvT, attnb,
                                          Bn, S, H, 64);
  }

  // 4) output projection (f32 out)
  {
    dim3 grid(Mrows / 64, D / 64);
    gemm_bias_kernel<0><<<grid, 128, 0, stream>>>(attnb, Wob, bo, (float*)d_out,
                                                  Mrows, D, D);
  }
}
